// CrossAttention_43688407335154
// MI455X (gfx1250) — matrix operand order repaired and checked by reading, not compiled, not run on hardware
//
#include <hip/hip_runtime.h>
#include <hip/hip_bf16.h>

typedef __attribute__((ext_vector_type(16))) _Float16 v16h;
typedef __attribute__((ext_vector_type(8)))  float    v8f;

#define BATCH 4
#define HEADS 16
#define SEQ   2048
#define CH    1024
#define HD    64
#define MROWS (BATCH * SEQ)   // 8192

// ------------------------------------------------ CDNA5 async-copy detection
#if defined(__has_builtin)
#if __has_builtin(__builtin_amdgcn_global_load_async_to_lds_b128)
#define HAVE_ASYNC 1
#endif
#endif
#ifndef HAVE_ASYNC
#define HAVE_ASYNC 0
#endif

#if HAVE_ASYNC
// Builtin signature (from hipcc diagnostic): (v4i __device__*, v4i __shared__*, Ii, Ii)
// with v4i = int __attribute__((vector_size(16))).
typedef int async_v4i __attribute__((vector_size(16)));
typedef async_v4i __attribute__((address_space(1)))* async_gp;
typedef async_v4i __attribute__((address_space(3)))* async_lp;

__device__ __forceinline__ void cp_async16(uint32_t* ldsDst, const void* gSrc) {
  __builtin_amdgcn_global_load_async_to_lds_b128(
      (async_gp)(unsigned long long)gSrc,
      (async_lp)(unsigned int)(unsigned long long)ldsDst,
      0, 0);
}
#endif

template <int N>
__device__ __forceinline__ void wait_async() {
#if HAVE_ASYNC
#if __has_builtin(__builtin_amdgcn_s_wait_asynccnt)
  __builtin_amdgcn_s_wait_asynccnt((short)N);
#else
  asm volatile("s_wait_asynccnt %0" :: "i"(N) : "memory");
#endif
#endif
}

// ---------------------------------------------------------------- helpers

__device__ __forceinline__ uint32_t pack2(float a, float b) {
  union { _Float16 h[2]; uint32_t u; } p;
  p.h[0] = (_Float16)a; p.h[1] = (_Float16)b;
  return p.u;
}

__device__ __forceinline__ v8f zero8() {
  v8f z;
  #pragma unroll
  for (int i = 0; i < 8; ++i) z[i] = 0.0f;
  return z;
}

// A-fragment (16x32 f16, MxK): lanes 0-15: M=lane, K=0..7 & 16..23; lanes 16-31: K=8..15 & 24..31.
__device__ __forceinline__ v16h frag_a(const uint32_t* lds, int rowBase, int ldPairs, int kPairOff) {
  int lane = threadIdx.x & 31;
  const uint32_t* p = lds + (size_t)(rowBase + (lane & 15)) * ldPairs + kPairOff + ((lane >> 4) << 2);
  union { v16h v; uint32_t u[8]; } f;
  #pragma unroll
  for (int g = 0; g < 4; ++g) { f.u[g] = p[g]; f.u[4 + g] = p[8 + g]; }
  return f.v;
}

// B-fragment (32x16 f16, KxN): lanes 0-15: N=lane, K=0..7 & 16..23; lanes 16-31: N=lane-16, K=8..15 & 24..31.
// LDS layout [n][k-pairs].
__device__ __forceinline__ v16h frag_b(const uint32_t* lds, int colBase, int ldPairs, int kPairOff) {
  int lane = threadIdx.x & 31;
  const uint32_t* p = lds + (size_t)(colBase + (lane & 15)) * ldPairs + kPairOff + ((lane >> 4) << 2);
  union { v16h v; uint32_t u[8]; } f;
  #pragma unroll
  for (int g = 0; g < 4; ++g) { f.u[g] = p[g]; f.u[4 + g] = p[8 + g]; }
  return f.v;
}

__device__ __forceinline__ v8f wmma16(v16h a, v16h b, v8f c) {
  return __builtin_amdgcn_wmma_f32_16x16x32_f16(false, a, false, b, (short)0, c, false, false);
}

// ---------------------------------------------------------------- QKV projection (+RoPE / +V-transpose)
// out = X @ W^T ; X:[M,CH] f32, W:[CH,CH] f32.  Double-buffered LDS, 1 barrier / K-step.
// mode 1: RoPE epilogue, out = f16 [B,H,SEQ,HD]  (Q or K)
// mode 0: transposed write, out = f16 [B,H,HD,SEQ] (V^T)
__global__ __launch_bounds__(256) void proj_qkv(const float* __restrict__ X,
                                                const float* __restrict__ W,
                                                _Float16* __restrict__ out,
                                                const int* __restrict__ pos,
                                                int mode) {
  __shared__ uint32_t ldsA[2][128 * 16];  // [row][kpair]
  __shared__ uint32_t ldsB[2][64 * 16];   // [n][kpair]

  const int tid  = threadIdx.x;
  const int lane = tid & 31;
  const int wave = tid >> 5;
  const int wm = wave >> 1, wn = wave & 1;          // 4x2 wave grid
  const int mBase = blockIdx.x * 128;
  const int nBase = blockIdx.y * 64;

  float4 fa[4], fb[2];
  auto load_regs = [&](int k0) {
    #pragma unroll
    for (int i = 0; i < 4; ++i) {
      int idx = tid + 256 * i; int row = idx >> 3, c4 = idx & 7;
      fa[i] = *(const float4*)(X + (size_t)(mBase + row) * CH + k0 + c4 * 4);
    }
    #pragma unroll
    for (int i = 0; i < 2; ++i) {
      int idx = tid + 256 * i; int row = idx >> 3, c4 = idx & 7;
      fb[i] = *(const float4*)(W + (size_t)(nBase + row) * CH + k0 + c4 * 4);
    }
  };
  auto store_lds = [&](int buf) {
    #pragma unroll
    for (int i = 0; i < 4; ++i) {
      int idx = tid + 256 * i; int row = idx >> 3, c4 = idx & 7;
      ldsA[buf][row * 16 + c4 * 2]     = pack2(fa[i].x, fa[i].y);
      ldsA[buf][row * 16 + c4 * 2 + 1] = pack2(fa[i].z, fa[i].w);
    }
    #pragma unroll
    for (int i = 0; i < 2; ++i) {
      int idx = tid + 256 * i; int row = idx >> 3, c4 = idx & 7;
      ldsB[buf][row * 16 + c4 * 2]     = pack2(fb[i].x, fb[i].y);
      ldsB[buf][row * 16 + c4 * 2 + 1] = pack2(fb[i].z, fb[i].w);
    }
  };

  v8f acc[2][2];
  #pragma unroll
  for (int i = 0; i < 2; ++i)
    #pragma unroll
    for (int j = 0; j < 2; ++j) acc[i][j] = zero8();

  load_regs(0);
  store_lds(0);
  int cur = 0;
  for (int k0 = 0; k0 < CH; k0 += 32) {
    const bool hasNext = (k0 + 32) < CH;
    if (hasNext) load_regs(k0 + 32);     // global loads overlap compute
    __syncthreads();                     // buf[cur] published, buf[cur^1] free

    v16h a0 = frag_a(ldsA[cur], wm * 32,      16, 0);
    v16h a1 = frag_a(ldsA[cur], wm * 32 + 16, 16, 0);
    v16h b0 = frag_b(ldsB[cur], wn * 32,      16, 0);
    v16h b1 = frag_b(ldsB[cur], wn * 32 + 16, 16, 0);
    acc[0][0] = wmma16(a0, b0, acc[0][0]);
    acc[0][1] = wmma16(a0, b1, acc[0][1]);
    acc[1][0] = wmma16(a1, b0, acc[1][0]);
    acc[1][1] = wmma16(a1, b1, acc[1][1]);

    if (hasNext) { store_lds(cur ^ 1); cur ^= 1; }
  }

  const int col = lane & 15;
  const int hi8 = (lane >> 4) << 3;

  if (mode == 0) {
    // V: write transposed [B,H,HD,SEQ]; 8 consecutive m per lane -> one b128 store per tile
    #pragma unroll
    for (int tm = 0; tm < 2; ++tm)
      #pragma unroll
      for (int tn = 0; tn < 2; ++tn) {
        int n = nBase + wn * 32 + tn * 16 + col;
        int h = n >> 6, d = n & 63;
        int m0 = mBase + wm * 32 + tm * 16 + hi8;
        int b = m0 >> 11, nk = m0 & (SEQ - 1);
        uint4 val;
        val.x = pack2(acc[tm][tn][0], acc[tm][tn][1]);
        val.y = pack2(acc[tm][tn][2], acc[tm][tn][3]);
        val.z = pack2(acc[tm][tn][4], acc[tm][tn][5]);
        val.w = pack2(acc[tm][tn][6], acc[tm][tn][7]);
        size_t idx = (((size_t)(b * HEADS + h) * HD + d) * SEQ + nk);
        *reinterpret_cast<uint4*>(out + idx) = val;
      }
  } else {
    // Q/K RoPE2D: wave's 32-wide aligned N window == one RoPE half; tile tn=0/tn=1 are
    // rotation partners with equal frequency (channel c and c+16 share inv_freq index).
    int cwin = nBase + wn * 32;
    int halfIdx = (cwin >> 5) & 1;                   // 0 -> y coord, 1 -> x coord
    float invf = __powf(100.0f, -(float)col * (1.0f / 16.0f));
    #pragma unroll
    for (int tm = 0; tm < 2; ++tm) {
      #pragma unroll
      for (int r = 0; r < 8; ++r) {
        int m = mBase + wm * 32 + tm * 16 + hi8 + r;
        int b = m >> 11, nq = m & (SEQ - 1);
        int p = pos[(size_t)(b * SEQ + nq) * 2 + halfIdx];
        float ang = (float)p * invf;
        float cs = __cosf(ang), sn = __sinf(ang);
        float a0 = acc[tm][0][r], a1 = acc[tm][1][r];
        float o0 = a0 * cs - a1 * sn;
        float o1 = a1 * cs + a0 * sn;
        int n0 = cwin + col;
        int h = n0 >> 6, d = n0 & 63;
        size_t base = ((size_t)(b * HEADS + h) * SEQ + nq) * HD;
        out[base + d]      = (_Float16)o0;
        out[base + d + 16] = (_Float16)o1;
      }
    }
  }
}

// ---------------------------------------------------------------- flash attention
// Q,K: f16 [B,H,SEQ,HD]; VT: f16 [B,H,HD,SEQ]; Xo: f16 [B,SEQ,CH]
// Double-buffered K/V tiles, async global->LDS copies (ASYNCcnt) when available.
__global__ __launch_bounds__(256) void attn_kernel(const _Float16* __restrict__ Q,
                                                   const _Float16* __restrict__ K,
                                                   const _Float16* __restrict__ VT,
                                                   _Float16* __restrict__ Xo) {
  __shared__ uint32_t ldsQ[128 * 32];      // [row][d-pair]    16 KB
  __shared__ uint32_t ldsK[2][64 * 32];    // [krow][d-pair]  2x8 KB
  __shared__ uint32_t ldsV[2][64 * 32];    // [d][krow-pair]  2x8 KB
  __shared__ uint32_t ldsP[128 * 32];      // [row][k-pair]    16 KB

  const int tid = threadIdx.x, wave = tid >> 5, lane = tid & 31;
  const int bh = blockIdx.y;                 // b*H + h
  const int qBase = blockIdx.x * 128;
  const float scale = 0.125f;                // D^-0.5, D=64

  const _Float16* Qh = Q  + ((size_t)bh * SEQ + qBase) * HD;
  const _Float16* Kh = K  + (size_t)bh * SEQ * HD;
  const _Float16* Vh = VT + (size_t)bh * HD * SEQ;

  // issue one 64-key K/V tile into buffer `buf` (4 async ops / thread, or sync fallback)
  auto issue_tile = [&](int buf, int kt) {
#if HAVE_ASYNC
    #pragma unroll
    for (int i = 0; i < 2; ++i) {
      int idx = tid + 256 * i;
      cp_async16(ldsK[buf] + idx * 4,
                 reinterpret_cast<const uint4*>(Kh + (size_t)kt * HD) + idx);
      int d = idx >> 3, c = idx & 7;
      cp_async16(ldsV[buf] + idx * 4,
                 reinterpret_cast<const uint4*>(Vh + (size_t)d * SEQ + kt + c * 8));
    }
#else
    #pragma unroll
    for (int i = 0; i < 2; ++i) {
      int idx = tid + 256 * i;
      reinterpret_cast<uint4*>(ldsK[buf])[idx] =
          reinterpret_cast<const uint4*>(Kh + (size_t)kt * HD)[idx];
      int d = idx >> 3, c = idx & 7;
      reinterpret_cast<uint4*>(ldsV[buf])[idx] =
          *reinterpret_cast<const uint4*>(Vh + (size_t)d * SEQ + kt + c * 8);
    }
#endif
  };

  // Q tile preload (contiguous 128*64 halves = 1024 uint4)
#if HAVE_ASYNC
  for (int i = tid; i < 1024; i += 256)
    cp_async16(ldsQ + i * 4, reinterpret_cast<const uint4*>(Qh) + i);
#else
  for (int i = tid; i < 1024; i += 256)
    reinterpret_cast<uint4*>(ldsQ)[i] = reinterpret_cast<const uint4*>(Qh)[i];
#endif
  issue_tile(0, 0);

  float mi[8], li[8];
  v8f Ofrag[4];
  #pragma unroll
  for (int r = 0; r < 8; ++r) { mi[r] = -3.0e38f; li[r] = 0.0f; }
  #pragma unroll
  for (int dt = 0; dt < 4; ++dt) Ofrag[dt] = zero8();

  _Float16* ldsPh = reinterpret_cast<_Float16*>(ldsP);

  int cur = 0;
  for (int kt = 0; kt < SEQ; kt += 64) {
    __syncthreads();                         // prior compute done; buf[cur^1] free
    const bool hasNext = (kt + 64) < SEQ;
    if (hasNext) { issue_tile(cur ^ 1, kt + 64); wait_async<4>(); }
    else         { wait_async<0>(); }
    __syncthreads();                         // tile `cur` visible to all waves

    // S = Q K^T  (this wave: q rows [wave*16, wave*16+16), all 64 keys)
    v16h aq0 = frag_a(ldsQ, wave * 16, 32, 0);
    v16h aq1 = frag_a(ldsQ, wave * 16, 32, 16);
    v8f S[4];
    #pragma unroll
    for (int nt = 0; nt < 4; ++nt) {
      v8f s = zero8();
      s = wmma16(aq0, frag_b(ldsK[cur], nt * 16, 32, 0),  s);
      s = wmma16(aq1, frag_b(ldsK[cur], nt * 16, 32, 16), s);
      S[nt] = s;
    }

    // online softmax (each row lives in one 16-lane half: shfl_xor 1/2/4/8)
    #pragma unroll
    for (int r = 0; r < 8; ++r) {
      float v0 = fmaxf(fmaxf(S[0][r], S[1][r]), fmaxf(S[2][r], S[3][r])) * scale;
      #pragma unroll
      for (int off = 1; off < 16; off <<= 1)
        v0 = fmaxf(v0, __shfl_xor(v0, off, 32));
      float mnew = fmaxf(mi[r], v0);
      float alpha = __expf(mi[r] - mnew);
      mi[r] = mnew;
      float rsum = 0.0f;
      #pragma unroll
      for (int nt = 0; nt < 4; ++nt) {
        float p = __expf(S[nt][r] * scale - mnew);
        S[nt][r] = p;
        rsum += p;
      }
      #pragma unroll
      for (int off = 1; off < 16; off <<= 1)
        rsum += __shfl_xor(rsum, off, 32);
      li[r] = li[r] * alpha + rsum;
      #pragma unroll
      for (int dt = 0; dt < 4; ++dt) Ofrag[dt][r] *= alpha;
    }

    // stage P (per-wave-private rows -> same-wave DS ordering, no barrier needed)
    {
      int row0 = wave * 16 + ((lane >> 4) << 3);
      int colw = lane & 15;
      #pragma unroll
      for (int nt = 0; nt < 4; ++nt)
        #pragma unroll
        for (int r = 0; r < 8; ++r)
          ldsPh[(size_t)(row0 + r) * 64 + nt * 16 + colw] = (_Float16)S[nt][r];
    }

    // O += P @ V
    v16h ap0 = frag_a(ldsP, wave * 16, 32, 0);
    v16h ap1 = frag_a(ldsP, wave * 16, 32, 16);
    #pragma unroll
    for (int dt = 0; dt < 4; ++dt) {
      Ofrag[dt] = wmma16(ap0, frag_b(ldsV[cur], dt * 16, 32, 0),  Ofrag[dt]);
      Ofrag[dt] = wmma16(ap1, frag_b(ldsV[cur], dt * 16, 32, 16), Ofrag[dt]);
    }
    cur ^= 1;
  }

  // finalize: O /= l, write f16 [B,SEQ,CH]
  const int b = bh >> 4, h = bh & 15;
  const int row0 = qBase + wave * 16 + ((lane >> 4) << 3);
  const int colw = lane & 15;
  #pragma unroll
  for (int dt = 0; dt < 4; ++dt) {
    int d = h * HD + dt * 16 + colw;
    #pragma unroll
    for (int r = 0; r < 8; ++r) {
      float o = Ofrag[dt][r] / li[r];
      Xo[((size_t)b * SEQ + row0 + r) * CH + d] = (_Float16)o;
    }
  }
}

// ---------------------------------------------------------------- output projection
// out = X @ Wo^T + bo ; X f16 [M,CH], Wo f32 [CH,CH], out f32.  Double-buffered.
__global__ __launch_bounds__(256) void proj_out(const _Float16* __restrict__ X,
                                                const float* __restrict__ W,
                                                const float* __restrict__ bias,
                                                float* __restrict__ out) {
  __shared__ uint32_t ldsA[2][128 * 16];
  __shared__ uint32_t ldsB[2][64 * 16];

  const int tid  = threadIdx.x;
  const int lane = tid & 31;
  const int wave = tid >> 5;
  const int wm = wave >> 1, wn = wave & 1;
  const int mBase = blockIdx.x * 128;
  const int nBase = blockIdx.y * 64;

  uint4  ra[2];
  float4 rb[2];
  auto load_regs = [&](int k0) {
    #pragma unroll
    for (int i = 0; i < 2; ++i) {
      int idx = tid + 256 * i; int row = idx >> 2, c = idx & 3;
      ra[i] = *reinterpret_cast<const uint4*>(X + (size_t)(mBase + row) * CH + k0 + c * 8);
    }
    #pragma unroll
    for (int i = 0; i < 2; ++i) {
      int idx = tid + 256 * i; int row = idx >> 3, c4 = idx & 7;
      rb[i] = *(const float4*)(W + (size_t)(nBase + row) * CH + k0 + c4 * 4);
    }
  };
  auto store_lds = [&](int buf) {
    #pragma unroll
    for (int i = 0; i < 2; ++i) {
      int idx = tid + 256 * i;
      reinterpret_cast<uint4*>(ldsA[buf])[idx] = ra[i];
    }
    #pragma unroll
    for (int i = 0; i < 2; ++i) {
      int idx = tid + 256 * i; int row = idx >> 3, c4 = idx & 7;
      ldsB[buf][row * 16 + c4 * 2]     = pack2(rb[i].x, rb[i].y);
      ldsB[buf][row * 16 + c4 * 2 + 1] = pack2(rb[i].z, rb[i].w);
    }
  };

  v8f acc[2][2];
  #pragma unroll
  for (int i = 0; i < 2; ++i)
    #pragma unroll
    for (int j = 0; j < 2; ++j) acc[i][j] = zero8();

  load_regs(0);
  store_lds(0);
  int cur = 0;
  for (int k0 = 0; k0 < CH; k0 += 32) {
    const bool hasNext = (k0 + 32) < CH;
    if (hasNext) load_regs(k0 + 32);
    __syncthreads();

    v16h a0 = frag_a(ldsA[cur], wm * 32,      16, 0);
    v16h a1 = frag_a(ldsA[cur], wm * 32 + 16, 16, 0);
    v16h b0 = frag_b(ldsB[cur], wn * 32,      16, 0);
    v16h b1 = frag_b(ldsB[cur], wn * 32 + 16, 16, 0);
    acc[0][0] = wmma16(a0, b0, acc[0][0]);
    acc[0][1] = wmma16(a0, b1, acc[0][1]);
    acc[1][0] = wmma16(a1, b0, acc[1][0]);
    acc[1][1] = wmma16(a1, b1, acc[1][1]);

    if (hasNext) { store_lds(cur ^ 1); cur ^= 1; }
  }

  const int colw = lane & 15;
  const int hi8 = (lane >> 4) << 3;
  #pragma unroll
  for (int tm = 0; tm < 2; ++tm)
    #pragma unroll
    for (int tn = 0; tn < 2; ++tn) {
      int n = nBase + wn * 32 + tn * 16 + colw;
      float bn = bias[n];
      int m0 = mBase + wm * 32 + tm * 16 + hi8;
      #pragma unroll
      for (int r = 0; r < 8; ++r)
        out[(size_t)(m0 + r) * CH + n] = acc[tm][tn][r] + bn;
    }
}

// ---------------------------------------------------------------- launch

extern "C" void kernel_launch(void* const* d_in, const int* in_sizes, int n_in,
                              void* d_out, int out_size, void* d_ws, size_t ws_size,
                              hipStream_t stream) {
  const float* query = (const float*)d_in[0];
  const float* key   = (const float*)d_in[1];
  const float* value = (const float*)d_in[2];
  const int*   qpos  = (const int*)d_in[3];
  const int*   kpos  = (const int*)d_in[4];
  const float* Wq    = (const float*)d_in[5];
  const float* Wk    = (const float*)d_in[6];
  const float* Wv    = (const float*)d_in[7];
  const float* Wo    = (const float*)d_in[8];
  const float* bo    = (const float*)d_in[9];
  float* out = (float*)d_out;

  const size_t elems = (size_t)BATCH * HEADS * SEQ * HD;   // 8M halves = 16 MB each
  _Float16* qws  = (_Float16*)d_ws;
  _Float16* kws  = qws  + elems;
  _Float16* vtws = kws  + elems;
  _Float16* xws  = vtws + elems;

  dim3 blk(256);
  dim3 gProj(MROWS / 128, CH / 64);          // 64 x 16
  proj_qkv<<<gProj, blk, 0, stream>>>(query, Wq, qws,  qpos, 1);
  proj_qkv<<<gProj, blk, 0, stream>>>(key,   Wk, kws,  kpos, 1);
  proj_qkv<<<gProj, blk, 0, stream>>>(value, Wv, vtws, nullptr, 0);

  dim3 gAttn(SEQ / 128, BATCH * HEADS);      // 16 x 64
  attn_kernel<<<gAttn, blk, 0, stream>>>(qws, kws, vtws, xws);

  proj_out<<<gProj, blk, 0, stream>>>(xws, Wo, bo, out);
}
